// LSTMModel_87600152969447
// MI455X (gfx1250) — matrix operand order repaired and checked by reading, not compiled, not run on hardware
//
#include <hip/hip_runtime.h>
#include <hip/hip_bf16.h>
#include <cstdint>

// ---------------------------------------------------------------------------
// CDNA5 (gfx1250) fused LSTM stack: bf16 WMMA (v_wmma_f32_16x16x32_bf16),
// fused input+recurrent GEMM  z = [x_t, h] @ [W;U] + b, gates + BN in LDS.
// ---------------------------------------------------------------------------

typedef __bf16 bf16_t;
typedef bf16_t v16bf __attribute__((ext_vector_type(16)));
typedef bf16_t v8bf  __attribute__((ext_vector_type(8)));
typedef float  v8f   __attribute__((ext_vector_type(8)));

union V16U { v16bf v; v8bf h[2]; };

__device__ __forceinline__ float sigmoid_f(float x) {
    return 1.0f / (1.0f + __expf(-x));
}

// ---------------------------------------------------------------------------
// Pack [W;U] (f32, row-major [Kc, 4H]) into bf16 WMMA B-fragment tile order:
// out[((nt*nKt + kt)*32 + lane)*16 + e]  <-  row k = kt*32 + 16*(lane/16) + e,
//                                            col n = nt*16 + (lane&15)
// so the scan kernel's per-lane B fragment is 16 contiguous bf16 (32 bytes).
// ---------------------------------------------------------------------------
__global__ __launch_bounds__(256) void pack_wu(
    const float* __restrict__ W,   // [F, 4H]
    const float* __restrict__ U,   // [H, 4H]
    bf16_t* __restrict__ out,
    int F, int N4, int nKt, int total)
{
    int idx = blockIdx.x * 256 + threadIdx.x;
    if (idx >= total) return;
    int e    = idx & 15;
    int lane = (idx >> 4) & 31;
    int kt   = (idx >> 9) % nKt;
    int nt   = idx / (512 * nKt);
    int k = kt * 32 + ((lane >> 4) << 4) + e;
    int n = nt * 16 + (lane & 15);
    float v = (k < F) ? W[(size_t)k * N4 + n] : U[(size_t)(k - F) * N4 + n];
    out[idx] = (bf16_t)v;
}

// ---------------------------------------------------------------------------
// Persistent-tile LSTM scan. Grid = B/16 workgroups, 512 threads (16 waves).
// Each WG owns 16 batch rows; loops over T timesteps.
//   A fragment (16x32 bf16): lane L holds row M=L%16; element e -> k-offset
//     kt*32 + 16*(L/16) + e   => two ds_read_b128 per k-step.
//   B fragment: packed contiguously (two global b128 loads per k-step).
//   C/D (16x16 f32): vgpr r, lane L -> M = r + 8*(L/16), N = L%16.
// ---------------------------------------------------------------------------
#define SCAN_THREADS 512
#define SCAN_WAVES   16

__global__ __launch_bounds__(SCAN_THREADS) void lstm_scan(
    const float*  __restrict__ in,    // [B, T, F]
    const bf16_t* __restrict__ wu,    // packed [Kc, 4H] bf16 (tile order)
    const float*  __restrict__ bias,  // [4H]
    const float*  __restrict__ bn_g,
    const float*  __restrict__ bn_b,
    const float*  __restrict__ bn_m,
    const float*  __restrict__ bn_v,
    float* __restrict__ out,          // seq_out: [B,T,H]  else [B,H] (t=T-1)
    int T, int F, int H, int seq_out)
{
    extern __shared__ char smem[];
    const int Kc  = F + H;
    const int N4  = 4 * H;
    const int nKt = Kc / 32;
    const int nNt = N4 / 16;

    float*  z_lds = (float*)smem;              // [16][4H]
    float*  c_lds = z_lds + 16 * N4;           // [16][H]
    bf16_t* a_lds = (bf16_t*)(c_lds + 16 * H); // [16][Kc]  (x_t | h), bf16

    const int tid  = threadIdx.x;
    const int lane = tid & 31;
    const int wid  = tid >> 5;
    const int hi   = lane >> 4;   // half-wave
    const int lo   = lane & 15;
    const int b0   = blockIdx.x * 16;

    // h = 0, c = 0
    for (int i = tid; i < 16 * H; i += SCAN_THREADS) {
        c_lds[i] = 0.0f;
        int r = i / H, j = i % H;
        a_lds[r * Kc + F + j] = (bf16_t)0.0f;
    }
    __syncthreads();

    for (int t = 0; t < T; ++t) {
        // ---- stage x_t into LDS (bf16) ----
        for (int i = tid; i < 16 * F; i += SCAN_THREADS) {
            int r = i / F, c = i % F;
            a_lds[r * Kc + c] =
                (bf16_t)in[((size_t)(b0 + r) * T + t) * F + c];
        }
        __syncthreads();

        // ---- z = [x_t, h] @ [W;U] + bias  (bf16 WMMA, f32 accum) ----
        const bf16_t* ap = a_lds + lo * Kc + 16 * hi;
        for (int nt = wid; nt < nNt; nt += SCAN_WAVES) {
            float bv = bias[nt * 16 + lo];
            v8f acc;
#pragma unroll
            for (int r = 0; r < 8; ++r) acc[r] = bv;

            const bf16_t* bp = wu + (size_t)nt * nKt * 512 + lane * 16;
            for (int kt = 0; kt < nKt; ++kt) {
                V16U a, b;
                a.h[0] = *(const v8bf*)(ap + kt * 32);
                a.h[1] = *(const v8bf*)(ap + kt * 32 + 8);
                const bf16_t* bl = bp + (size_t)kt * 512;
                b.h[0] = *(const v8bf*)(bl);
                b.h[1] = *(const v8bf*)(bl + 8);
                __builtin_prefetch(bl + 512, 0, 1);  // next k-tile into L2
                acc = __builtin_amdgcn_wmma_f32_16x16x32_bf16(
                    false, a.v, false, b.v, (short)0, acc, false, false);
            }
#pragma unroll
            for (int r = 0; r < 8; ++r)
                z_lds[(r + 8 * hi) * N4 + nt * 16 + lo] = acc[r];
        }
        __syncthreads();

        // ---- gates, cell update, leaky + BN, write-out ----
        for (int i = tid; i < 16 * H; i += SCAN_THREADS) {
            int r = i / H, j = i % H;
            float zi = z_lds[r * N4 + j];
            float zf = z_lds[r * N4 + H + j];
            float zg = z_lds[r * N4 + 2 * H + j];
            float zo = z_lds[r * N4 + 3 * H + j];
            float ig = sigmoid_f(zi);
            float fg = sigmoid_f(zf);
            float og = sigmoid_f(zo);
            float gg = zg > 0.0f ? zg : 0.0f;
            float c  = fg * c_lds[i] + ig * gg;
            c_lds[i] = c;
            float h = og * (c > 0.0f ? c : 0.0f);
            a_lds[r * Kc + F + j] = (bf16_t)h;   // recurrent input for t+1

            float y = h >= 0.0f ? h : 0.1f * h;  // leaky
            y = bn_g[j] * (y - bn_m[j]) * rsqrtf(bn_v[j] + 1e-3f) + bn_b[j];
            if (seq_out) {
                out[((size_t)(b0 + r) * T + t) * H + j] = y;
            } else if (t == T - 1) {
                out[(size_t)(b0 + r) * H + j] = y;
            }
        }
        __syncthreads();
    }
}

// ---------------------------------------------------------------------------
// Final dense: out[b] = h[b,:] . dW + db
// ---------------------------------------------------------------------------
__global__ __launch_bounds__(256) void dense_out(
    const float* __restrict__ h,   // [B, 32]
    const float* __restrict__ dW,  // [32]
    const float* __restrict__ db,  // [1]
    float* __restrict__ out, int B, int H)
{
    int b = blockIdx.x * 256 + threadIdx.x;
    if (b >= B) return;
    float s = db[0];
#pragma unroll
    for (int j = 0; j < 32; ++j) s += h[(size_t)b * H + j] * dW[j];
    out[b] = s;
}

// ---------------------------------------------------------------------------
extern "C" void kernel_launch(void* const* d_in, const int* in_sizes, int n_in,
                              void* d_out, int out_size, void* d_ws, size_t ws_size,
                              hipStream_t stream)
{
    const float* x    = (const float*)d_in[0];
    const float* l1W  = (const float*)d_in[1];
    const float* l1U  = (const float*)d_in[2];
    const float* l1b  = (const float*)d_in[3];
    const float* l2W  = (const float*)d_in[4];
    const float* l2U  = (const float*)d_in[5];
    const float* l2b  = (const float*)d_in[6];
    const float* l3W  = (const float*)d_in[7];
    const float* l3U  = (const float*)d_in[8];
    const float* l3b  = (const float*)d_in[9];
    const float* bn1g = (const float*)d_in[10];
    const float* bn1b = (const float*)d_in[11];
    const float* bn1m = (const float*)d_in[12];
    const float* bn1v = (const float*)d_in[13];
    const float* bn2g = (const float*)d_in[14];
    const float* bn2b = (const float*)d_in[15];
    const float* bn2m = (const float*)d_in[16];
    const float* bn2v = (const float*)d_in[17];
    const float* bn3g = (const float*)d_in[18];
    const float* bn3b = (const float*)d_in[19];
    const float* bn3m = (const float*)d_in[20];
    const float* bn3v = (const float*)d_in[21];
    const float* dW   = (const float*)d_in[22];
    const float* db   = (const float*)d_in[23];

    const int B = 256, T = 512;

    // workspace carve (256B aligned)
    char* ws = (char*)d_ws;
    size_t off = 0;
    auto carve = [&](size_t bytes) -> void* {
        void* p = ws + off;
        off = (off + bytes + 255) & ~(size_t)255;
        return p;
    };
    bf16_t* pk1 = (bf16_t*)carve((size_t)384 * 1280 * 2);
    bf16_t* pk2 = (bf16_t*)carve((size_t)512 * 768 * 2);
    bf16_t* pk3 = (bf16_t*)carve((size_t)224 * 128 * 2);
    float*  h1  = (float*)carve((size_t)B * T * 320 * 4);
    float*  h2  = (float*)carve((size_t)B * T * 192 * 4);
    float*  hl  = (float*)carve((size_t)B * 32 * 4);

    // pack [W;U] -> bf16 WMMA-fragment layout (L2-resident afterwards)
    { int tot = 384 * 1280; pack_wu<<<(tot + 255) / 256, 256, 0, stream>>>(l1W, l1U, pk1,  64, 1280, 12, tot); }
    { int tot = 512 * 768;  pack_wu<<<(tot + 255) / 256, 256, 0, stream>>>(l2W, l2U, pk2, 320,  768, 16, tot); }
    { int tot = 224 * 128;  pack_wu<<<(tot + 255) / 256, 256, 0, stream>>>(l3W, l3U, pk3, 192,  128,  7, tot); }

    // layer 1: F=64, H=320
    size_t lds1 = (size_t)(16 * 1280 + 16 * 320) * 4 + (size_t)16 * 384 * 2;
    lstm_scan<<<B / 16, SCAN_THREADS, lds1, stream>>>(
        x, pk1, l1b, bn1g, bn1b, bn1m, bn1v, h1, T, 64, 320, 1);

    // layer 2: F=320, H=192
    size_t lds2 = (size_t)(16 * 768 + 16 * 192) * 4 + (size_t)16 * 512 * 2;
    lstm_scan<<<B / 16, SCAN_THREADS, lds2, stream>>>(
        h1, pk2, l2b, bn2g, bn2b, bn2m, bn2v, h2, T, 320, 192, 1);

    // layer 3: F=192, H=32, last step only
    size_t lds3 = (size_t)(16 * 128 + 16 * 32) * 4 + (size_t)16 * 224 * 2;
    lstm_scan<<<B / 16, SCAN_THREADS, lds3, stream>>>(
        h2, pk3, l3b, bn3g, bn3b, bn3m, bn3v, hl, T, 192, 32, 0);

    dense_out<<<(B + 255) / 256, 256, 0, stream>>>(hl, dW, db, (float*)d_out, B, 32);
}
